// AutoregressiveRNN_24601572671595
// MI455X (gfx1250) — matrix operand order repaired and checked by reading, not compiled, not run on hardware
//
#include <hip/hip_runtime.h>

// ---------------------------------------------------------------------------
// GRU + post-linear for MI455X (gfx1250), wave32, bf16 WMMA w/ f32 accumulate.
// ---------------------------------------------------------------------------

typedef __bf16 bf16;
typedef __attribute__((ext_vector_type(16))) __bf16 v16bf;
typedef __attribute__((ext_vector_type(8)))  __bf16 v8bf;
typedef __attribute__((ext_vector_type(8)))  float  v8f;

#define B_  32
#define T_  1024
#define C_  128
#define G_  896
#define HD  896
#define OUT_HS_ELEMS (B_ * T_ * HD)   // 29360128 floats, then h_last tail

// ---- WMMA helpers ---------------------------------------------------------

__device__ __forceinline__ v8f wmma_bf16(v16bf a, v16bf b, v8f c) {
  // D = A(16x32 bf16) x B(32x16 bf16) + C(16x16 f32)
  return __builtin_amdgcn_wmma_f32_16x16x32_bf16(
      /*neg_a=*/false, a, /*neg_b=*/false, b,
      /*c_mod=*/(short)0, c, /*reuse_a=*/false, /*reuse_b=*/false);
}

// A fragment: row-major A, tile origin already applied (row 0, k 0 of tile).
// Layout (matches B fragment): lanes 0-15 hold K[0..15] for row M=lane,
// lanes 16-31 hold K[16..31] for row M=lane-16.
__device__ __forceinline__ v16bf load_a_frag(const bf16* __restrict__ A,
                                             int lda, int lane) {
  int khalf = (lane >> 4) << 4;                 // 0 or 16
  const bf16* p = A + (lane & 15) * lda + khalf;
  v8bf a0 = *(const v8bf*)(p);                  // K + khalf + [0..7]
  v8bf a1 = *(const v8bf*)(p + 8);              // K + khalf + [8..15]
  return __builtin_shufflevector(a0, a1,
      0,1,2,3,4,5,6,7, 8,9,10,11,12,13,14,15);
}

// B fragment: W stored row-major [N_total, K_total] (i.e. torch weight [out,in]),
// so B(k,n) = W[n][k].  Layout used here, same as A (16-bit B 32x16): lane = n + 16*(k>=16),
// 16 consecutive K values packed 2-per-VGPR -> one contiguous 32-byte load.
__device__ __forceinline__ v16bf load_b_frag(const bf16* __restrict__ W,
                                             int ldw, int lane) {
  const bf16* p = W + (lane & 15) * ldw + ((lane >> 4) << 4);
  return *(const v16bf*)(p);
}

__device__ __forceinline__ float sigmoidf_(float x) {
  return 1.0f / (1.0f + __expf(-x));
}

// ---- prep kernels ---------------------------------------------------------

__global__ void convert_f32_bf16_kernel(const float* __restrict__ src,
                                        bf16* __restrict__ dst, int n) {
  int i = blockIdx.x * blockDim.x + threadIdx.x;
  if (i < n) dst[i] = (bf16)src[i];
}

__global__ void init_h_kernel(const float* __restrict__ state,
                              float* __restrict__ hf,
                              bf16* __restrict__ hb, int n) {
  int i = blockIdx.x * blockDim.x + threadIdx.x;
  if (i < n) { float v = state[i]; hf[i] = v; hb[i] = (bf16)v; }
}

// ---- GRU step kernel (launched T times; stream order = scan dependency) ----
// Grid: 56 blocks (one per 16-column tile of G). Block: 192 threads = 6 waves.
// Wave w -> gate g = w/2 (r,z,n), batch-half mh = w&1 (rows 0-15 / 16-31).
// Each wave: acc_h = h·w_hh tile (28 WMMAs), acc_x = x_t·w_ih tile (4 WMMAs).
__global__ __launch_bounds__(192)
void gru_step_kernel(const bf16* __restrict__ xb,    // [B, T, C] bf16
                     const bf16* __restrict__ wihb,  // [3G, C]  bf16
                     const bf16* __restrict__ whhb,  // [3G, G]  bf16
                     const float* __restrict__ b_ih, // [3G]
                     const float* __restrict__ b_hh, // [3G]
                     const bf16* __restrict__ h_src_b,  // [B, G] bf16
                     const float* __restrict__ h_src_f, // [B, G] f32
                     bf16* __restrict__ h_dst_b,
                     float* __restrict__ h_dst_f,
                     bf16* __restrict__ hs_b,        // [B, T, G] bf16
                     float* __restrict__ hlast_out,  // non-null only at t=T-1
                     int t)
{
  // planes: 0=r(h+x), 1=z(h+x), 2=hn (h-side of n), 3=xn (x-side of n)
  __shared__ float lds[4][B_][16];

  const int tid  = threadIdx.x;
  const int wave = tid >> 5;
  const int lane = tid & 31;
  const int g    = wave >> 1;        // 0..2
  const int mh   = wave & 1;         // 0..1
  const int j0   = blockIdx.x * 16;

  v8f acc_h = {};
  v8f acc_x = {};

  // recurrent GEMM: A = h_src_b rows [mh*16 .. mh*16+15], K = G
  const bf16* Abase = h_src_b + mh * 16 * G_;
  const bf16* Wbase = whhb + (g * G_ + j0) * G_;
  #pragma unroll 4
  for (int k = 0; k < G_; k += 32) {
    v16bf a = load_a_frag(Abase + k, G_, lane);
    v16bf b = load_b_frag(Wbase + k, G_, lane);
    acc_h = wmma_bf16(a, b, acc_h);
  }

  // input GEMM: A = x[:, t, :] rows [mh*16 ..], row stride T*C, K = C
  const bf16* Xbase  = xb + (mh * 16) * (T_ * C_) + t * C_;
  const bf16* Wibase = wihb + (g * G_ + j0) * C_;
  #pragma unroll
  for (int k = 0; k < C_; k += 32) {
    v16bf a = load_a_frag(Xbase + k, T_ * C_, lane);
    v16bf b = load_b_frag(Wibase + k, C_, lane);
    acc_x = wmma_bf16(a, b, acc_x);
  }

  // spill to LDS. r/z gates can fuse h+x; n gate keeps them separate.
  const int jj    = lane & 15;
  const int mbase = mh * 16 + ((lane >> 4) << 3);
  if (g < 2) {
    v8f s = acc_h + acc_x;
    #pragma unroll
    for (int i = 0; i < 8; ++i) lds[g][mbase + i][jj] = s[i];
  } else {
    #pragma unroll
    for (int i = 0; i < 8; ++i) {
      lds[2][mbase + i][jj] = acc_h[i];
      lds[3][mbase + i][jj] = acc_x[i];
    }
  }
  __syncthreads();

  // gate math on 32x16 tile (512 elems, 192 threads)
  for (int idx = tid; idx < B_ * 16; idx += 192) {
    int b  = idx >> 4;
    int jl = idx & 15;
    int j  = j0 + jl;
    float r  = sigmoidf_(lds[0][b][jl] + b_ih[j]       + b_hh[j]);
    float z  = sigmoidf_(lds[1][b][jl] + b_ih[G_ + j]  + b_hh[G_ + j]);
    float hn = lds[2][b][jl] + b_hh[2 * G_ + j];
    float xn = lds[3][b][jl] + b_ih[2 * G_ + j];
    float n  = tanhf(xn + r * hn);
    float h_old = h_src_f[b * G_ + j];
    float h_new = (1.0f - z) * n + z * h_old;

    h_dst_f[b * G_ + j] = h_new;
    h_dst_b[b * G_ + j] = (bf16)h_new;
    hs_b[(b * T_ + t) * G_ + j] = (bf16)h_new;
    if (hlast_out) hlast_out[b * G_ + j] = h_new;
  }
}

// ---- post GEMM: out = relu(hs · w_postᵀ + b_post) -------------------------
// M = B*T = 32768, N = 896, K = 896.  64x64 block tile, 8 waves, 2 acc/wave.
__global__ __launch_bounds__(256)
void post_gemm_kernel(const bf16* __restrict__ A,    // [32768, 896] hs bf16
                      const bf16* __restrict__ W,    // [896, 896] w_post bf16
                      const float* __restrict__ bias,
                      float* __restrict__ out)       // [32768, 896] f32
{
  const int tid  = threadIdx.x;
  const int wave = tid >> 5;
  const int lane = tid & 31;
  const int mi   = wave & 3;
  const int ni0  = (wave >> 2) << 1;
  const int m0   = blockIdx.x * 64 + mi * 16;
  const int n0   = blockIdx.y * 64 + ni0 * 16;

  v8f acc0 = {};
  v8f acc1 = {};
  const bf16* Ab = A + m0 * G_;
  const bf16* W0 = W + n0 * G_;
  const bf16* W1 = W + (n0 + 16) * G_;
  #pragma unroll 4
  for (int k = 0; k < G_; k += 32) {
    v16bf a  = load_a_frag(Ab + k, G_, lane);
    v16bf b0 = load_b_frag(W0 + k, G_, lane);
    v16bf b1 = load_b_frag(W1 + k, G_, lane);
    acc0 = wmma_bf16(a, b0, acc0);
    acc1 = wmma_bf16(a, b1, acc1);
  }

  const int col0  = n0 + (lane & 15);
  const int col1  = col0 + 16;
  const int rbase = m0 + ((lane >> 4) << 3);
  const float bb0 = bias[col0];
  const float bb1 = bias[col1];
  #pragma unroll
  for (int i = 0; i < 8; ++i) {
    out[(size_t)(rbase + i) * HD + col0] = fmaxf(acc0[i] + bb0, 0.0f);
    out[(size_t)(rbase + i) * HD + col1] = fmaxf(acc1[i] + bb1, 0.0f);
  }
}

// ---- host launch ----------------------------------------------------------

extern "C" void kernel_launch(void* const* d_in, const int* in_sizes, int n_in,
                              void* d_out, int out_size, void* d_ws, size_t ws_size,
                              hipStream_t stream) {
  (void)in_sizes; (void)n_in; (void)out_size; (void)ws_size;
  const float* x      = (const float*)d_in[0];  // [B,T,C]
  const float* state  = (const float*)d_in[1];  // [1,B,G]
  const float* w_ih   = (const float*)d_in[2];  // [3G,C]
  const float* w_hh   = (const float*)d_in[3];  // [3G,G]
  const float* b_ih   = (const float*)d_in[4];  // [3G]
  const float* b_hh   = (const float*)d_in[5];  // [3G]
  const float* w_post = (const float*)d_in[6];  // [HD,G]
  const float* b_post = (const float*)d_in[7];  // [HD]
  float* out = (float*)d_out;

  // workspace carve-out (all 256B aligned)
  char* ws = (char*)d_ws;
  size_t off = 0;
  auto carve = [&](size_t bytes) -> void* {
    void* p = ws + off;
    off += (bytes + 255) & ~(size_t)255;
    return p;
  };
  const int n_x    = B_ * T_ * C_;        // 4194304
  const int n_wih  = 3 * G_ * C_;         // 344064
  const int n_whh  = 3 * G_ * G_;         // 2408448
  const int n_wpo  = HD * G_;             // 802816
  const int n_h    = B_ * G_;             // 28672
  const int n_hs   = B_ * T_ * G_;        // 29360128

  bf16*  xb     = (bf16*)carve((size_t)n_x   * 2);
  bf16*  wihb   = (bf16*)carve((size_t)n_wih * 2);
  bf16*  whhb   = (bf16*)carve((size_t)n_whh * 2);
  bf16*  wpostb = (bf16*)carve((size_t)n_wpo * 2);
  bf16*  hs_b   = (bf16*)carve((size_t)n_hs  * 2);
  float* hf0    = (float*)carve((size_t)n_h * 4);
  float* hf1    = (float*)carve((size_t)n_h * 4);
  bf16*  hb0    = (bf16*)carve((size_t)n_h * 2);
  bf16*  hb1    = (bf16*)carve((size_t)n_h * 2);

  // 1) convert inputs/weights to bf16
  convert_f32_bf16_kernel<<<(n_x   + 255) / 256, 256, 0, stream>>>(x, xb, n_x);
  convert_f32_bf16_kernel<<<(n_wih + 255) / 256, 256, 0, stream>>>(w_ih, wihb, n_wih);
  convert_f32_bf16_kernel<<<(n_whh + 255) / 256, 256, 0, stream>>>(w_hh, whhb, n_whh);
  convert_f32_bf16_kernel<<<(n_wpo + 255) / 256, 256, 0, stream>>>(w_post, wpostb, n_wpo);
  init_h_kernel<<<(n_h + 255) / 256, 256, 0, stream>>>(state, hf0, hb0, n_h);

  // 2) sequential GRU scan: T step kernels, stream-ordered, ping-pong h
  float* hf[2] = {hf0, hf1};
  bf16*  hb[2] = {hb0, hb1};
  for (int t = 0; t < T_; ++t) {
    int s = t & 1;
    gru_step_kernel<<<G_ / 16, 192, 0, stream>>>(
        xb, wihb, whhb, b_ih, b_hh,
        hb[s], hf[s], hb[1 - s], hf[1 - s],
        hs_b, (t == T_ - 1) ? (out + OUT_HS_ELEMS) : nullptr, t);
  }

  // 3) post linear + relu -> d_out
  post_gemm_kernel<<<dim3((B_ * T_) / 64, HD / 64), 256, 0, stream>>>(
      hs_b, wpostb, b_post, out);
}
